// GeodesicLoss_25829933318525
// MI455X (gfx1250) — matrix operand order repaired and checked by reading, not compiled, not run on hardware
//
#include <hip/hip_runtime.h>
#include <cstdint>
#include <cstddef>

// ---------------------------------------------------------------------------
// GeodesicLoss for MI455X (gfx1250): B=256, D=64, 10 RK2 steps.
// Strategy: convert Christoffel tensor to bf16 once (134 MB, fits 192MB L2;
// f32 source streamed with NT hint so it doesn't evict the bf16 copy), then
// run the 20 sequential  a_i = -G_b(64x4096) . (v (x) v)  contractions with
// v_wmma_f32_16x16x32_bf16, one workgroup per batch row.
// ---------------------------------------------------------------------------

typedef __attribute__((ext_vector_type(16))) __bf16 v16bf;
typedef __attribute__((ext_vector_type(8)))  __bf16 v8bf;
typedef __attribute__((ext_vector_type(4)))  __bf16 v4bf;
typedef __attribute__((ext_vector_type(8)))  float  v8f;
typedef __attribute__((ext_vector_type(16))) float  v16f;
typedef __attribute__((ext_vector_type(4)))  float  v4f;

#define GB_B     256
#define GB_D     64
#define GB_K     (GB_D * GB_D)   // 4096
#define GB_STEPS 10
#define GB_DT    (1.0f / GB_STEPS)

// ---------------- workspace layout ----------------
// floats[0..255]    : per-block euclid partials  ||x0-x1||
// floats[256..511]  : per-block target-penalty partials
// floats[512..767]  : per-block smoothness partials
// uint  @ floats[768]: "gamma has any nonzero" flag
// byte offset 4096  : bf16 gamma copy (256*64*4096 * 2B = 128 MiB)

// ---------------- helpers ----------------

__device__ __forceinline__ void reduce64(float* red, int tid) {
  __syncthreads();
  if (tid < 32) red[tid] += red[tid + 32];
  __syncthreads();
  if (tid < 16) red[tid] += red[tid + 16];
  __syncthreads();
  if (tid <  8) red[tid] += red[tid +  8];
  __syncthreads();
  if (tid <  4) red[tid] += red[tid +  4];
  __syncthreads();
  if (tid <  2) red[tid] += red[tid +  2];
  __syncthreads();
  if (tid <  1) red[tid] += red[tid +  1];
  __syncthreads();
}

// B-operand loaders: 16 contiguous values of one gamma row.
__device__ __forceinline__ v16bf load_bfrag(const __bf16* p) {
  return *(const v16bf*)p;                     // 32 B -> 2x global_load_b128
}
__device__ __forceinline__ v16bf load_bfrag(const float* p) {
  v16f g = *(const v16f*)p;                    // 64 B -> 4x global_load_b128
  v16bf o;
#pragma unroll
  for (int r = 0; r < 16; ++r) o[r] = (__bf16)g[r];
  return o;
}

// One accel evaluation: acc = -einsum('ijk,j,k->i', G_b, vin, vin).
// Must be called by all 256 threads (contains barriers, WMMA needs full EXEC).
template <typename GT>
__device__ __forceinline__ void accel_eval(const GT* __restrict__ gptr,
                                           const float* __restrict__ vin,
                                           __bf16* __restrict__ wlds,
                                           float (*partial)[16],
                                           float* __restrict__ acc,
                                           int tid, int wave, int half,
                                           int nn, int kgrp) {
  // ---- build w = vin (outer) vin in bf16, 4096 entries, 16 per thread ----
  {
    const int   j  = tid >> 2;            // each 16-run shares one j
    const int   kb = (tid & 3) * 16;
    const float vj = vin[j];
    v8bf lo, hi;
#pragma unroll
    for (int r = 0; r < 8; ++r) lo[r] = (__bf16)(vj * vin[kb + r]);
#pragma unroll
    for (int r = 0; r < 8; ++r) hi[r] = (__bf16)(vj * vin[kb + 8 + r]);
    *(v8bf*)(wlds + tid * 16)     = lo;
    *(v8bf*)(wlds + tid * 16 + 8) = hi;
  }
  __syncthreads();

  // ---- WMMA contraction: this wave covers i-tile (wave&3), K half kgrp ----
  v8f c = {0.f, 0.f, 0.f, 0.f, 0.f, 0.f, 0.f, 0.f};
  const int kofs = kgrp * (GB_K / 2);
#pragma unroll 8
  for (int cc = 0; cc < 64; ++cc) {
    const int cbase = kofs + cc * 32;
    // B fragment: lane nn holds gamma[i0+nn][cbase + half*16 .. +16)
    v16bf bmat = load_bfrag(gptr + cbase);
    // A fragment (replicated w): lanes 0-15 need K=0..15; 16-31: K=16..31
    v8bf a0 = *(const v8bf*)(wlds + cbase + half * 16);
    v8bf a1 = *(const v8bf*)(wlds + cbase + half * 16 + 8);
    v16bf amat = __builtin_shufflevector(a0, a1, 0, 1, 2, 3, 4, 5, 6, 7,
                                         8, 9, 10, 11, 12, 13, 14, 15);
    c = __builtin_amdgcn_wmma_f32_16x16x32_bf16(
        /*neg_a=*/false, amat, /*neg_b=*/false, bmat,
        /*c_mod=*/(short)0, c, /*reuse_a=*/false, /*reuse_b=*/false);
  }
  // All D rows identical (A rows replicated): a_partial[i0+nn] = c[0], lanes 0-15.
  if (half == 0) partial[wave][nn] = c[0];
  __syncthreads();
  if (tid < GB_D)
    acc[tid] = -(partial[tid >> 4][tid & 15] + partial[(tid >> 4) + 4][tid & 15]);
  __syncthreads();
}

// ---------------- kernels ----------------

__global__ void init_kernel(unsigned* flag) { *flag = 0u; }

// f32 -> bf16 conversion (optional) + any-nonzero flag. 4 floats per thread.
// NT load: the one-shot 268 MB f32 stream must not evict the L2-resident
// bf16 copy being produced.
template <bool STORE>
__global__ __launch_bounds__(256) void gamma_prep(const float* __restrict__ g,
                                                  __bf16* __restrict__ out,
                                                  unsigned* __restrict__ flag) {
  const size_t idx = ((size_t)blockIdx.x * 256 + threadIdx.x) * 4;
  const v4f v = __builtin_nontemporal_load((const v4f*)(g + idx));
  if (STORE) {
    v4bf o = {(__bf16)v[0], (__bf16)v[1], (__bf16)v[2], (__bf16)v[3]};
    *(v4bf*)(out + idx) = o;   // regular-temporal store: keep in L2
  }
  __shared__ int anynz;
  if (threadIdx.x == 0) anynz = 0;
  __syncthreads();
  if (v[0] != 0.f || v[1] != 0.f || v[2] != 0.f || v[3] != 0.f) anynz = 1;
  __syncthreads();
  if (threadIdx.x == 0 && anynz) atomicOr(flag, 1u);
}

template <typename GT>
__global__ __launch_bounds__(256) void geodesic_kernel(
    const float* __restrict__ x0g, const float* __restrict__ x1g,
    const GT* __restrict__ gamma, float* __restrict__ partEu,
    float* __restrict__ partTp, float* __restrict__ partSm) {
  __shared__ float  x1s[GB_D];
  __shared__ float  vel[GB_D];
  __shared__ float  vin[GB_D];
  __shared__ float  acc[GB_D];
  __shared__ float  red[GB_D];
  __shared__ float  hist[GB_STEPS + 1][GB_D];
  __shared__ __bf16 wlds[GB_K];
  __shared__ float  partial[8][16];

  const int tid  = threadIdx.x;
  const int b    = blockIdx.x;
  const int wave = tid >> 5, lane = tid & 31;
  const int half = lane >> 4, nn = lane & 15;
  const int tile = wave & 3, kgrp = wave >> 2;

  // Per-lane base: gamma row (i0 + nn) of batch b, shifted by K-half of lane group.
  const GT* gptr =
      gamma + ((size_t)(b * GB_D + tile * 16 + nn)) * GB_K + half * 16;

  if (tid < GB_D) {
    const float a = x0g[b * GB_D + tid];
    const float t = x1g[b * GB_D + tid];
    x1s[tid]     = t;
    hist[0][tid] = a;
    const float d = (t - a) * GB_DT;
    vel[tid] = d;
    red[tid] = d * d;
  }
  reduce64(red, tid);
  const float dnorm = sqrtf(red[0]);        // ||(x1-x0)/S||
  if (tid == 0) partEu[b] = dnorm * (float)GB_STEPS;  // ||x1-x0||
  if (tid < GB_D) vel[tid] = vel[tid] / (dnorm + 1e-8f);
  __syncthreads();

  // ---- RK2 integration, 10 steps, 2 accel evals each ----
  for (int s = 0; s < GB_STEPS; ++s) {
    if (tid < GB_D) vin[tid] = vel[tid];
    __syncthreads();
    accel_eval(gptr, vin, wlds, partial, acc, tid, wave, half, nn, kgrp);

    if (tid < GB_D) vin[tid] = vel[tid] + (0.5f * GB_DT) * acc[tid];
    __syncthreads();
    accel_eval(gptr, vin, wlds, partial, acc, tid, wave, half, nn, kgrp);

    if (tid < GB_D) {
      const float nv = vel[tid] + GB_DT * acc[tid];
      vel[tid]        = nv;
      hist[s + 1][tid] = hist[s][tid] + nv * GB_DT;
    }
    __syncthreads();
  }

  // ---- target penalty: ||traj_final - x1|| ----
  if (tid < GB_D) {
    const float d = hist[GB_STEPS][tid] - x1s[tid];
    red[tid] = d * d;
  }
  reduce64(red, tid);
  if (tid == 0) partTp[b] = sqrtf(red[0]);

  // ---- smoothness: sum_s ||path[s+1] - 2 path[s] + path[s-1]|| ----
  float smacc = 0.f;
  for (int s = 1; s < GB_STEPS; ++s) {
    __syncthreads();
    if (tid < GB_D) {
      const float sec = hist[s + 1][tid] - 2.f * hist[s][tid] + hist[s - 1][tid];
      red[tid] = sec * sec;
    }
    reduce64(red, tid);
    if (tid == 0) smacc += sqrtf(red[0]);
  }
  if (tid == 0) partSm[b] = smacc;
}

__global__ __launch_bounds__(256) void finalize_kernel(
    const float* __restrict__ partEu, const float* __restrict__ partTp,
    const float* __restrict__ partSm, const unsigned* __restrict__ flag,
    float* __restrict__ out) {
  __shared__ float r0[256], r1[256], r2[256];
  const int tid = threadIdx.x;
  r0[tid] = partEu[tid];
  r1[tid] = partTp[tid];
  r2[tid] = partSm[tid];
  for (int off = 128; off > 0; off >>= 1) {
    __syncthreads();
    if (tid < off) {
      r0[tid] += r0[tid + off];
      r1[tid] += r1[tid + off];
      r2[tid] += r2[tid + off];
    }
  }
  __syncthreads();
  if (tid == 0) {
    const float euclid = r0[0] * (1.0f / GB_B);
    const float geo =
        r1[0] * (1.0f / GB_B) + 0.1f * (r2[0] * (1.0f / GB_B));
    out[0] = (*flag) ? geo : euclid;
  }
}

// ---------------- host entry ----------------

extern "C" void kernel_launch(void* const* d_in, const int* in_sizes, int n_in,
                              void* d_out, int out_size, void* d_ws,
                              size_t ws_size, hipStream_t stream) {
  const float* x0    = (const float*)d_in[0];  // outputs  [256,64] f32
  const float* x1    = (const float*)d_in[1];  // targets  [256,64] f32
  const float* gamma = (const float*)d_in[2];  // Christoffel [256,64,64,64] f32

  float*    ws     = (float*)d_ws;
  float*    partEu = ws;
  float*    partTp = ws + 256;
  float*    partSm = ws + 512;
  unsigned* flag   = (unsigned*)(ws + 768);
  __bf16*   gbf    = (__bf16*)((char*)d_ws + 4096);

  const size_t nG   = (size_t)GB_B * GB_D * GB_K;       // 67,108,864
  const size_t need = 4096 + nG * 2;                    // ~128 MiB

  init_kernel<<<1, 1, 0, stream>>>(flag);

  const int prepBlocks = (int)(nG / (256 * 4));         // 65536
  if (ws_size >= need) {
    gamma_prep<true><<<prepBlocks, 256, 0, stream>>>(gamma, gbf, flag);
    geodesic_kernel<__bf16><<<GB_B, 256, 0, stream>>>(x0, x1, gbf, partEu,
                                                      partTp, partSm);
  } else {
    gamma_prep<false><<<prepBlocks, 256, 0, stream>>>(gamma, nullptr, flag);
    geodesic_kernel<float><<<GB_B, 256, 0, stream>>>(x0, x1, gamma, partEu,
                                                     partTp, partSm);
  }

  finalize_kernel<<<1, 256, 0, stream>>>(partEu, partTp, partSm, flag,
                                         (float*)d_out);
}
